// MixtralBLockSparseTop2MLP_HQQ_88287347736657
// MI455X (gfx1250) — compile-verified, matrix operand order checked by reading, not run on hardware
//
#include <hip/hip_runtime.h>
#include <stdint.h>

typedef __attribute__((ext_vector_type(16))) _Float16 v16h;
typedef __attribute__((ext_vector_type(8)))  _Float16 v8h;
typedef __attribute__((ext_vector_type(4)))  _Float16 v4h;
typedef __attribute__((ext_vector_type(8)))  float    v8f;
typedef __attribute__((ext_vector_type(4)))  int      v4i;

#define T_DIM 256
#define H_DIM 4096
#define I_DIM 14336
#define GSZ   64
#define G1    (H_DIM / GSZ)   /* 64  */
#define G2    (I_DIM / GSZ)   /* 224 */

#define CTILE 32              /* output-column strip per block        */
#define SROW  (GSZ + 8)       /* padded LDS row stride (72 halfs)     */

// Workspace layout: [x16: T*H f16 = 2 MB][hbuf: T*I f16 = 7.34 MB]
#define X16_ELEMS ((long)T_DIM * H_DIM)

// ---------------------------------------------------------------------------
// Async global->LDS copy (16B per lane), ASYNCcnt-tracked; bypasses VGPRs.
// Builtin signature (from clang diagnostic): (int4 AS1*, int4 AS3*, imm, imm).
// Address-space casts go through uintptr_t (generic global == AS1 numerically;
// generic LDS truncated to 32 bits == AS3 offset per the aperture rules).
// ---------------------------------------------------------------------------
__device__ __forceinline__ void async_cp_b128(_Float16* dst, const _Float16* src) {
#if defined(__has_builtin) && __has_builtin(__builtin_amdgcn_global_load_async_to_lds_b128)
  typedef __attribute__((address_space(1))) v4i* gp_t;
  typedef __attribute__((address_space(3))) v4i* lp_t;
  __builtin_amdgcn_global_load_async_to_lds_b128(
      (gp_t)(uintptr_t)src, (lp_t)(uintptr_t)dst, 0, 0);
#else
  const uint32_t l = (uint32_t)(uintptr_t)dst;  // low 32 bits of generic = LDS byte addr
  asm volatile("global_load_async_to_lds_b128 %0, %1, off"
               :: "v"(l), "v"(src) : "memory");
#endif
}

__device__ __forceinline__ void wait_async0() {
#if defined(__has_builtin) && __has_builtin(__builtin_amdgcn_s_wait_asynccnt)
  __builtin_amdgcn_s_wait_asynccnt(0);
#else
  asm volatile("s_wait_asynccnt 0x0" ::: "memory");
#endif
}

// ---------------------------------------------------------------------------
// WMMA 16-bit operand layout (ISA 7.12.2): lane = hsel*16 + idx,
//   idx = A-row (M) / B-col (N);
//   v16h elems 0..7  <- K = koff + {0..7},  elems 8..15 <- K = koff+16+{0..7}
//   with koff = kstep*32 + hsel*8.
// ---------------------------------------------------------------------------
__device__ __forceinline__ v16h lds_frag(const _Float16* __restrict__ row, int koff) {
  const v8h lo = *(const v8h*)(row + koff);
  const v8h hi = *(const v8h*)(row + koff + 16);
  v16h a;
#pragma unroll
  for (int e = 0; e < 8; ++e) { a[e] = lo[e]; a[e + 8] = hi[e]; }
  return a;
}

// HQQ dequant of 8 consecutive q values: w = q*sc + bias (bias = -zero*sc),
// written as one 16B chunk into LDS.
__device__ __forceinline__ void dequant8_store(_Float16* __restrict__ dst,
                                               const int* __restrict__ src,
                                               float sc, float bias) {
  const int4 q0 = *(const int4*)(src);
  const int4 q1 = *(const int4*)(src + 4);
  v8h o;
  o[0] = (_Float16)fmaf((float)q0.x, sc, bias);
  o[1] = (_Float16)fmaf((float)q0.y, sc, bias);
  o[2] = (_Float16)fmaf((float)q0.z, sc, bias);
  o[3] = (_Float16)fmaf((float)q0.w, sc, bias);
  o[4] = (_Float16)fmaf((float)q1.x, sc, bias);
  o[5] = (_Float16)fmaf((float)q1.y, sc, bias);
  o[6] = (_Float16)fmaf((float)q1.z, sc, bias);
  o[7] = (_Float16)fmaf((float)q1.w, sc, bias);
  *(v8h*)dst = o;
}

// ---------------------------------------------------------------------------
// Kernel 0: cast x (f32) -> x16 (f16) once, so GEMM A-loads are pure f16.
// ---------------------------------------------------------------------------
__global__ __launch_bounds__(256) void hqq_xcast(const float* __restrict__ x,
                                                 _Float16* __restrict__ x16, int n) {
  const int i = (blockIdx.x * blockDim.x + threadIdx.x) * 4;
  if (i < n) {
    const float4 v = *(const float4*)(x + i);
    v4h h;
    h[0] = (_Float16)v.x; h[1] = (_Float16)v.y;
    h[2] = (_Float16)v.z; h[3] = (_Float16)v.w;
    *(v4h*)(x16 + i) = h;
  }
}

// ---------------------------------------------------------------------------
// Kernel 1: h = silu(x@w1^T) * (x@w3^T)  -> hbuf (f16)
// Block: 256 thr = 8 waves, covers [T=256 x I=32]. Waves: 2 col-tiles x 4
// row-tiles (4 sub-accumulators each). Weights dequantized to LDS exactly
// once; activation K-panel async-copied to LDS once per block.
// ---------------------------------------------------------------------------
__global__ __launch_bounds__(256) void hqq_gateup_wmma(
    const _Float16* __restrict__ x16,
    const int* __restrict__ w1q, const float* __restrict__ w1s, const float* __restrict__ w1z,
    const int* __restrict__ w3q, const float* __restrict__ w3s, const float* __restrict__ w3z,
    _Float16* __restrict__ hbuf) {
  __shared__ _Float16 sA[T_DIM * SROW];   // 256x72 halfs = 36 KB
  __shared__ _Float16 sB1[CTILE * SROW];  // 32x72 halfs = 4.5 KB
  __shared__ _Float16 sB3[CTILE * SROW];

  const int tid   = threadIdx.x;
  const int lane  = tid & 31;
  const int wave  = tid >> 5;
  const int hsel  = lane >> 4;
  const int idx   = lane & 15;
  const int waveC = wave & 1;    // column sub-tile (16 cols)
  const int waveT = wave >> 1;   // row sub-tile base (64 rows)

  const int iBase = blockIdx.x * CTILE;

  // staging roles: 8 threads per weight row, one 8-elem chunk each
  const int wrow = tid >> 3;          // 0..31
  const int wchk = (tid & 7) * 8;     // 0,8,..,56
  const int irow = iBase + wrow;
  const int*   q1p = w1q + (long)irow * H_DIM;
  const int*   q3p = w3q + (long)irow * H_DIM;
  const float* s1p = w1s + (long)irow * G1;
  const float* z1p = w1z + (long)irow * G1;
  const float* s3p = w3s + (long)irow * G1;
  const float* z3p = w3z + (long)irow * G1;
  const _Float16* xrow = x16 + (long)tid * H_DIM;  // thread copies its T-row

  v8f accG[4], accU[4];
#pragma unroll
  for (int tt = 0; tt < 4; ++tt) { accG[tt] = (v8f){}; accU[tt] = (v8f){}; }

#pragma unroll 1
  for (int g = 0; g < G1; ++g) {
    const int k0 = g * GSZ;
    // ---- stage activation K-panel via async global->LDS (no VGPR traffic) ----
#pragma unroll
    for (int c = 0; c < 8; ++c) {
      async_cp_b128(&sA[tid * SROW + c * 8], xrow + k0 + c * 8);
    }
    // ---- stage weights (dequant exactly once) ----
    {
      const float s1 = s1p[g]; const float b1 = -z1p[g] * s1;
      const float s3 = s3p[g]; const float b3 = -z3p[g] * s3;
      dequant8_store(&sB1[wrow * SROW + wchk], q1p + k0 + wchk, s1, b1);
      dequant8_store(&sB3[wrow * SROW + wchk], q3p + k0 + wchk, s3, b3);
    }
    if (g + 1 < G1) {
      __builtin_prefetch(q1p + k0 + GSZ + wchk, 0, 1);
      __builtin_prefetch(q3p + k0 + GSZ + wchk, 0, 1);
    }
    wait_async0();
    __syncthreads();
    // ---- WMMA over the staged K=64 panel ----
#pragma unroll
    for (int kk = 0; kk < 2; ++kk) {
      const int  koff = kk * 32 + hsel * 8;
      const v16h fb1  = lds_frag(&sB1[(waveC * 16 + idx) * SROW], koff);
      const v16h fb3  = lds_frag(&sB3[(waveC * 16 + idx) * SROW], koff);
#pragma unroll
      for (int tt = 0; tt < 4; ++tt) {
        const v16h fa = lds_frag(&sA[(waveT * 64 + tt * 16 + idx) * SROW], koff);
        accG[tt] = __builtin_amdgcn_wmma_f32_16x16x32_f16(false, fa, false, fb1,
                                                          (short)0, accG[tt], false, false);
        accU[tt] = __builtin_amdgcn_wmma_f32_16x16x32_f16(false, fa, false, fb3,
                                                          (short)0, accU[tt], false, false);
      }
    }
    __syncthreads();
  }

  // C/D layout: VGPR r, lane -> (M = r + 8*hsel, N = idx)
  const int icol = iBase + waveC * 16 + idx;
#pragma unroll
  for (int tt = 0; tt < 4; ++tt) {
#pragma unroll
    for (int r = 0; r < 8; ++r) {
      const float gv = accG[tt][r];
      const float uv = accU[tt][r];
      const float sg = gv / (1.0f + __expf(-gv));  // silu(g)
      const int   t  = waveT * 64 + tt * 16 + r + hsel * 8;
      hbuf[(long)t * I_DIM + icol] = (_Float16)(sg * uv);
    }
  }
}

// ---------------------------------------------------------------------------
// Kernel 2: out = h @ w2^T (f32 out). Same scheme: block [T=256 x H=32].
// ---------------------------------------------------------------------------
__global__ __launch_bounds__(256) void hqq_down_wmma(
    const _Float16* __restrict__ hbuf,
    const int* __restrict__ w2q, const float* __restrict__ w2s, const float* __restrict__ w2z,
    float* __restrict__ out) {
  __shared__ _Float16 sA[T_DIM * SROW];  // 36 KB
  __shared__ _Float16 sB[CTILE * SROW];  // 4.5 KB

  const int tid   = threadIdx.x;
  const int lane  = tid & 31;
  const int wave  = tid >> 5;
  const int hsel  = lane >> 4;
  const int idx   = lane & 15;
  const int waveC = wave & 1;
  const int waveT = wave >> 1;

  const int nBase = blockIdx.x * CTILE;

  const int wrow = tid >> 3;
  const int wchk = (tid & 7) * 8;
  const int nrow = nBase + wrow;
  const int*   q2p = w2q + (long)nrow * I_DIM;
  const float* s2p = w2s + (long)nrow * G2;
  const float* z2p = w2z + (long)nrow * G2;
  const _Float16* hrow = hbuf + (long)tid * I_DIM;

  v8f acc[4];
#pragma unroll
  for (int tt = 0; tt < 4; ++tt) acc[tt] = (v8f){};

#pragma unroll 1
  for (int g = 0; g < G2; ++g) {
    const int k0 = g * GSZ;
#pragma unroll
    for (int c = 0; c < 8; ++c) {
      async_cp_b128(&sA[tid * SROW + c * 8], hrow + k0 + c * 8);
    }
    {
      const float s2 = s2p[g]; const float b2 = -z2p[g] * s2;
      dequant8_store(&sB[wrow * SROW + wchk], q2p + k0 + wchk, s2, b2);
    }
    if (g + 1 < G2) __builtin_prefetch(q2p + k0 + GSZ + wchk, 0, 1);
    wait_async0();
    __syncthreads();
#pragma unroll
    for (int kk = 0; kk < 2; ++kk) {
      const int  koff = kk * 32 + hsel * 8;
      const v16h fb   = lds_frag(&sB[(waveC * 16 + idx) * SROW], koff);
#pragma unroll
      for (int tt = 0; tt < 4; ++tt) {
        const v16h fa = lds_frag(&sA[(waveT * 64 + tt * 16 + idx) * SROW], koff);
        acc[tt] = __builtin_amdgcn_wmma_f32_16x16x32_f16(false, fa, false, fb,
                                                         (short)0, acc[tt], false, false);
      }
    }
    __syncthreads();
  }

  const int ncol = nBase + waveC * 16 + idx;
#pragma unroll
  for (int tt = 0; tt < 4; ++tt) {
#pragma unroll
    for (int r = 0; r < 8; ++r) {
      const int t = waveT * 64 + tt * 16 + r + hsel * 8;
      out[(long)t * H_DIM + ncol] = acc[tt][r];
    }
  }
}

// ---------------------------------------------------------------------------
extern "C" void kernel_launch(void* const* d_in, const int* in_sizes, int n_in,
                              void* d_out, int out_size, void* d_ws, size_t ws_size,
                              hipStream_t stream) {
  (void)in_sizes; (void)n_in; (void)out_size; (void)ws_size;

  const float* x   = (const float*)d_in[0];
  const int*   w1q = (const int*)  d_in[1];
  const float* w1s = (const float*)d_in[2];
  const float* w1z = (const float*)d_in[3];
  const int*   w3q = (const int*)  d_in[4];
  const float* w3s = (const float*)d_in[5];
  const float* w3z = (const float*)d_in[6];
  const int*   w2q = (const int*)  d_in[7];
  const float* w2s = (const float*)d_in[8];
  const float* w2z = (const float*)d_in[9];
  float*       out = (float*)d_out;

  _Float16* x16  = (_Float16*)d_ws;
  _Float16* hbuf = x16 + X16_ELEMS;

  // K0: x f32 -> f16
  {
    const int n = T_DIM * H_DIM;
    hqq_xcast<<<dim3(n / (256 * 4)), 256, 0, stream>>>(x, x16, n);
  }
  // K1: gate/up + silu -> hbuf   grid: I/32 = 448 blocks
  hqq_gateup_wmma<<<dim3(I_DIM / CTILE), 256, 0, stream>>>(
      x16, w1q, w1s, w1z, w3q, w3s, w3z, hbuf);
  // K2: down-proj -> out         grid: H/32 = 128 blocks
  hqq_down_wmma<<<dim3(H_DIM / CTILE), 256, 0, stream>>>(
      hbuf, w2q, w2s, w2z, out);
}
